// ED_Fourth_MOE_36326833389789
// MI455X (gfx1250) — matrix operand order repaired and checked by reading, not compiled, not run on hardware
//
#include <hip/hip_runtime.h>
#include <hip/hip_bf16.h>

typedef __attribute__((ext_vector_type(16))) _Float16 v16h;
typedef __attribute__((ext_vector_type(8)))  _Float16 v8h;
typedef __attribute__((ext_vector_type(8)))  float    v8f;
typedef __attribute__((ext_vector_type(2)))  float    v2f;

#define NB   1024   // batch tokens
#define TT   10     // time steps
#define NH   32     // H = W
#define ND   1024   // token dim
#define NHID 2048   // expert hidden
#define NE   4      // experts
#define NMOE 5
#define MAXT 36     // max 32-row tiles per moe (sum ceil(c_e/32) <= 36)

// ---------------- workspace layout (bytes) ----------------
static const size_t O_TOK   = 0;                      // B*D f32      = 4 MB
static const size_t O_TOKH  = 4194304;                // B*D f16      = 2 MB
static const size_t O_W1F   = 6291456;                // frags f16    = 80 MB
static const size_t O_W2F   = 90177536;               // frags f16    = 80 MB
static const size_t O_MOUT  = 174063616;              // 5*B*D f32    = 20 MB
static const size_t O_SEL   = 195035136;              // 5*B int
static const size_t O_TIDL  = 195055616;              // 5*B int
static const size_t O_CNT   = 195076096;              // 20 int
static const size_t O_OFF   = O_CNT + 80;             // 20 int
static const size_t O_CUR   = O_CNT + 160;            // 20 int
static const size_t O_NBLK  = O_CNT + 240;            // 5 int (+pad)
static const size_t O_TILES = O_CNT + 264;            // 5*36 int2

// conv LDS: sIn 13*20*36*4 = 37440 | sW1 1080 | sW2 3600 | dump | sO1 259200
static const int CONV_SMEM = 301376;
static const int GEMM_SMEM = 196736;   // sA 64KB + sH 128KB + sTid 128B

// ---------------- WMMA helpers ----------------
__device__ inline v8f wmma_f16(v16h a, v16h b, v8f c) {
  return __builtin_amdgcn_wmma_f32_16x16x32_f16(false, a, false, b, (short)0, c, false, false);
}
// B fragment: prepped layout, 16 contiguous f16 per lane
__device__ inline v16h ldB(const _Float16* p) {
  v8h lo = *(const v8h*)p;
  v8h hi = *(const v8h*)(p + 8);
  v16h r;
#pragma unroll
  for (int i = 0; i < 8; ++i) { r[i] = lo[i]; r[i + 8] = hi[i]; }
  return r;
}
// A fragment from row-major f16 (LDS): p = rowbase + kt*32 + hf*8; second half at +16
__device__ inline v16h ldA(const _Float16* p) {
  v8h lo = *(const v8h*)p;
  v8h hi = *(const v8h*)(p + 16);
  v16h r;
#pragma unroll
  for (int i = 0; i < 8; ++i) { r[i] = lo[i]; r[i + 8] = hi[i]; }
  return r;
}

// ---------------- weight -> f16 WMMA-fragment repack ----------------
// dst layout: ((((mat*NT + nt)*KT + kt)*32 + lane)*16 + sub)
// element (k,n) of a KxN row-major matrix: lane = (n%16) + 16*((k%16)/8), sub = (k%8) + 8*((k%32)/16)
__global__ __launch_bounds__(256) void prep_frags(const float* __restrict__ src,
                                                  _Float16* __restrict__ dst,
                                                  int K, int N, int KT, int NTl) {
  unsigned t = blockIdx.x * 256u + threadIdx.x;
  unsigned u = t >> 1, hs = t & 1;
  unsigned lane = u & 31u;
  unsigned f = u >> 5;
  unsigned kt = f % KT; f /= KT;
  unsigned nt = f % NTl;
  unsigned mat = f / NTl;
  unsigned n = nt * 16 + (lane & 15);
  unsigned kb = kt * 32 + (lane >> 4) * 8 + hs * 16;
  union { _Float16 h[8]; float4 q; } uo;
#pragma unroll
  for (int s = 0; s < 8; ++s)
    uo.h[s] = (_Float16)src[((size_t)mat * K + (kb + s)) * N + n];
  size_t base = ((((size_t)mat * NTl + nt) * KT + kt) * 32 + lane) * 16 + hs * 8;
  *(float4*)(dst + base) = uo.q;
}

// ---------------- fused conv1(+relu) + conv2(+relu) per (image, half) ----------------
#if defined(__gfx1250__) && __has_builtin(__builtin_amdgcn_wmma_f32_16x16x4_f32)
#define CONV_WMMA 1
#endif

__global__ __launch_bounds__(256) void conv_fused_k(const float* __restrict__ x,
                                                    const float* __restrict__ cw1,
                                                    const float* __restrict__ cb1,
                                                    const float* __restrict__ cw2,
                                                    const float* __restrict__ cb2,
                                                    float* __restrict__ tok,
                                                    _Float16* __restrict__ tokh) {
  extern __shared__ char smem[];
  float* sIn   = (float*)smem;                 // [13][20][36]: t-rows 0,11,12 zero pad
  float* sW1   = (float*)(smem + 37440);       // 270
  float* sW2   = (float*)(smem + 38520);       // 900
  float* sDump = (float*)(smem + 42120);       // 1 scratch word for pad-channel stores
  float* sO1   = (float*)(smem + 42176);       // [10][10][18][36] conv1 output (+zero halo)

  int b = blockIdx.x >> 1, half = blockIdx.x & 1;
  int hbase = half * 16;
  int tid = threadIdx.x;

  for (int i = tid; i < 270; i += 256) sW1[i] = cw1[i];
  for (int i = tid; i < 900; i += 256) sW2[i] = cw2[i];
  {
    float4 z4 = make_float4(0.f, 0.f, 0.f, 0.f);
    float4* z = (float4*)sO1;
    for (int i = tid; i < 16200; i += 256) z[i] = z4;
  }
  // input tile with zero halo in t (row = t+1), h (row hbase-2+r), w (col = w+1)
  for (int i = tid; i < 13 * 20 * 36; i += 256) {
    int ts = i / 720, r = (i / 36) % 20, c = i % 36;
    int tin = ts - 1;
    int hh = hbase - 2 + r;
    int w = c - 1;
    float v = 0.f;
    if (tin >= 0 && tin < 10 && hh >= 0 && hh < 32 && w >= 0 && w < 32)
      v = x[(((size_t)b * TT + tin) * NH + hh) * NH + w];
    sIn[i] = v;
  }
  __syncthreads();

  // ---- conv1: out1[c][t][h][w] via f32 WMMA (M=channel pad 10->16, K=27->28, N=16 w-positions)
#ifdef CONV_WMMA
  {
    int wid = tid >> 5, lane = tid & 31;
    int cc = lane & 15, hf = lane >> 4;
    v2f Aw[7];
    int ofs0[7], ofs1[7];
    float cbr[8];
#pragma unroll
    for (int kt = 0; kt < 7; ++kt) {
      float w0 = 0.f, w1v = 0.f;
      int k0 = kt * 4 + 2 * hf;
      if (cc < 10 && k0     < 27) w0  = sW1[cc * 27 + k0];
      if (cc < 10 && k0 + 1 < 27) w1v = sW1[cc * 27 + k0 + 1];
      v2f tmp = {w0, w1v};
      Aw[kt] = tmp;
      // address offsets; k>=27 has zero A-weight so clamp address (finite garbage * 0 = 0)
      int ka = (k0     > 26) ? 26 : k0;
      int kb2 = (k0 + 1 > 26) ? 26 : k0 + 1;
      ofs0[kt] = (ka / 9) * 720 + ((ka % 9) / 3) * 36 + (ka % 3);
      ofs1[kt] = (kb2 / 9) * 720 + ((kb2 % 9) / 3) * 36 + (kb2 % 3);
    }
#pragma unroll
    for (int r = 0; r < 8; ++r) {
      int c2 = r + 8 * hf;
      cbr[r] = (c2 < 10) ? cb1[c2] : 0.f;
    }
    // valid conv1 rows only: half0 -> hr 1..17, half1 -> hr 0..16 (17 rows); others stay zero
    int hr_lo = (half == 0) ? 1 : 0;
    for (int tile = wid; tile < 340; tile += 8) {
      int wt = tile & 1;
      int tmpi = tile >> 1;
      int t = tmpi / 17;
      int hr = hr_lo + (tmpi % 17);
      int wcol = wt * 16 + cc;
      const float* basep = sIn + (t * 20 + hr) * 36 + wcol;
      v8f C = {};
#pragma unroll
      for (int kt = 0; kt < 7; ++kt) {
        float b0 = basep[ofs0[kt]];
        float b1v = basep[ofs1[kt]];
        v2f Bv = {b0, b1v};
        C = __builtin_amdgcn_wmma_f32_16x16x4_f32(false, Aw[kt], false, Bv, (short)0, C, false, false);
      }
      // unconditional stores: pad channels (c2>=10) redirected to dump word (addr select,
      // no EXEC branches; multiple lanes hitting the same LDS word is fine)
#pragma unroll
      for (int r = 0; r < 8; ++r) {
        int c2 = r + 8 * hf;
        float v = C[r] + cbr[r];
        v = v > 0.f ? v : 0.f;
        float* dst = (c2 < 10) ? (sO1 + ((c2 * 10 + t) * 18 + hr) * 36 + wcol + 1) : sDump;
        *dst = v;
      }
    }
  }
#else
  for (int o = tid; o < 10 * 10 * 18 * 32; o += 256) {
    int w = o & 31;
    int tmpi = o >> 5;
    int hr = tmpi % 18; tmpi /= 18;
    int t = tmpi % 10, c = tmpi / 10;
    int hh = hbase - 1 + hr;
    if (hh < 0 || hh > 31) continue;
    float acc = cb1[c];
#pragma unroll
    for (int dt = 0; dt < 3; ++dt)
#pragma unroll
      for (int dh = 0; dh < 3; ++dh)
#pragma unroll
        for (int dw = 0; dw < 3; ++dw)
          acc += sIn[((t + dt) * 20 + hr + dh) * 36 + w + dw] * sW1[c * 27 + dt * 9 + dh * 3 + dw];
    acc = acc > 0.f ? acc : 0.f;
    sO1[((c * 10 + t) * 18 + hr) * 36 + w + 1] = acc;
  }
#endif
  __syncthreads();

  // ---- conv2: 1 output channel GEMV (900-term), 64 threads, strip-of-4 + paired halves
  if (tid < 64) {
    int ty = tid >> 2, wq = tid & 3;
    int wx4 = wq * 4;
    float bias = cb2[0];
    float ax[4] = {bias, bias, bias, bias};
    float ay[4] = {bias, bias, bias, bias};
    for (int c = 0; c < 10; ++c)
      for (int t = 0; t < 10; ++t) {
        const float* wrow = &sW2[(c * 10 + t) * 9];
#pragma unroll
        for (int dh = 0; dh < 3; ++dh) {
          const float* row = &sO1[((c * 10 + t) * 18 + ty + dh) * 36];
          float4 qa0 = *(const float4*)(row + wx4);
          float4 qa1 = *(const float4*)(row + wx4 + 4);
          float4 qb0 = *(const float4*)(row + wx4 + 16);
          float4 qb1 = *(const float4*)(row + wx4 + 20);
          float rA[8] = {qa0.x, qa0.y, qa0.z, qa0.w, qa1.x, qa1.y, qa1.z, qa1.w};
          float rB[8] = {qb0.x, qb0.y, qb0.z, qb0.w, qb1.x, qb1.y, qb1.z, qb1.w};
#pragma unroll
          for (int dw = 0; dw < 3; ++dw) {
            float wv = wrow[dh * 3 + dw];
#pragma unroll
            for (int j = 0; j < 4; ++j) {
              ax[j] += wv * rA[j + dw];
              ay[j] += wv * rB[j + dw];
            }
          }
        }
      }
    int h = hbase + ty;
#pragma unroll
    for (int j = 0; j < 4; ++j) {
      float v1 = ax[j] > 0.f ? ax[j] : 0.f;
      float v2 = ay[j] > 0.f ? ay[j] : 0.f;
      size_t i1 = (size_t)b * ND + h * NH + wx4 + j;
      tok[i1] = v1;        tokh[i1] = (_Float16)v1;
      tok[i1 + 16] = v2;   tokh[i1 + 16] = (_Float16)v2;
    }
  }
}

// ---------------- gating: top-1 argmax over 4 experts, wave32 reduction ----------------
__global__ __launch_bounds__(256) void gating_k(const float* __restrict__ tok,
                                                const float* __restrict__ wg,
                                                int* __restrict__ sel,
                                                int* __restrict__ counts) {
  int moe = blockIdx.y;
  int b = blockIdx.x * 8 + (threadIdx.x >> 5);
  int lane = threadIdx.x & 31;
  const float* tr = tok + (size_t)b * ND;
  const float4* wg4 = (const float4*)(wg + (size_t)moe * ND * NE);
  float a0 = 0.f, a1 = 0.f, a2 = 0.f, a3 = 0.f;
  for (int i = 0; i < 32; ++i) {
    int j = i * 32 + lane;
    float tv = tr[j];
    float4 g = wg4[j];
    a0 += tv * g.x; a1 += tv * g.y; a2 += tv * g.z; a3 += tv * g.w;
  }
#pragma unroll
  for (int off = 16; off >= 1; off >>= 1) {
    a0 += __shfl_xor(a0, off, 32);
    a1 += __shfl_xor(a1, off, 32);
    a2 += __shfl_xor(a2, off, 32);
    a3 += __shfl_xor(a3, off, 32);
  }
  if (lane == 0) {
    int best = 0; float bv = a0;
    if (a1 > bv) { bv = a1; best = 1; }
    if (a2 > bv) { bv = a2; best = 2; }
    if (a3 > bv) { bv = a3; best = 3; }
    sel[moe * NB + b] = best;
    atomicAdd(&counts[moe * NE + best], 1);
  }
}

// ---------------- per-moe bucket offsets + M-tile table ----------------
__global__ void setup_k(const int* __restrict__ counts, int* __restrict__ offs,
                        int2* __restrict__ tiles, int* __restrict__ nblk) {
  if (threadIdx.x) return;
  int moe = blockIdx.x;
  int o = 0, nb = 0;
  for (int e = 0; e < NE; ++e) {
    int c = counts[moe * NE + e];
    offs[moe * NE + e] = o;
    for (int s = 0; s < c; s += 32) tiles[moe * MAXT + nb++] = make_int2(e, o + s);
    o += c;
  }
  nblk[moe] = nb;
}

__global__ __launch_bounds__(256) void scatter_k(const int* __restrict__ sel,
                                                 const int* __restrict__ offs,
                                                 int* __restrict__ cursors,
                                                 int* __restrict__ tidl) {
  int moe = blockIdx.y;
  int b = blockIdx.x * 256 + threadIdx.x;
  int e = sel[moe * NB + b];
  int pos = atomicAdd(&cursors[moe * NE + e], 1);
  tidl[moe * NB + offs[moe * NE + e] + pos] = b;
}

// importance == load == counts when k=1 => loss = sum 2e-2 * var(counts,ddof=1)/(mean^2+eps)
__global__ void loss_k(const int* __restrict__ counts, float* __restrict__ out, int outn) {
  if (threadIdx.x || blockIdx.x) return;
  float total = 0.f;
  for (int m = 0; m < NMOE; ++m) {
    float mean = (float)NB / NE;
    float var = 0.f;
    for (int e = 0; e < NE; ++e) {
      float d = (float)counts[m * NE + e] - mean;
      var += d * d;
    }
    var /= (NE - 1);
    total += 2.f * 1e-2f * (var / (mean * mean + 1e-10f));
  }
  out[outn - 1] = total;
}

// ---------------- grouped MoE expert MLP: 32-token tile, full N, f16 WMMA ----------------
__global__ __launch_bounds__(256) void moe_gemm_k(const _Float16* __restrict__ tokh,
                                                  const _Float16* __restrict__ w1f,
                                                  const _Float16* __restrict__ w2f,
                                                  const float* __restrict__ b1,
                                                  const float* __restrict__ b2,
                                                  const int* __restrict__ tidl,
                                                  const int* __restrict__ counts,
                                                  const int* __restrict__ offs,
                                                  const int2* __restrict__ tiles,
                                                  const int* __restrict__ nblk,
                                                  float* __restrict__ mout) {
  extern __shared__ char smem[];
  _Float16* sA = (_Float16*)smem;                    // 32 x 1024 f16 (gathered tokens)
  _Float16* sH = (_Float16*)(smem + 65536);          // 32 x 2048 f16 (hidden)
  int* sTid = (int*)(smem + 196608);                 // 32 token ids (-1 = pad)

  int moe = blockIdx.y;
  if ((int)blockIdx.x >= nblk[moe]) return;
  int2 tl = tiles[moe * MAXT + blockIdx.x];
  int e = tl.x, rs = tl.y;
  int lim = offs[moe * NE + e] + counts[moe * NE + e];
  int tid = threadIdx.x;

  {  // stage 32 gathered token rows into LDS (zero pad rows)
    int rowl = tid >> 3, seg = tid & 7;
    int grow = rs + rowl;
    float4* dst = (float4*)(sA + rowl * ND + seg * 128);
    if (grow < lim) {
      const float4* srcp = (const float4*)(tokh + (size_t)tidl[moe * NB + grow] * ND + seg * 128);
#pragma unroll
      for (int i = 0; i < 16; ++i) dst[i] = srcp[i];
    } else {
      float4 z = make_float4(0.f, 0.f, 0.f, 0.f);
#pragma unroll
      for (int i = 0; i < 16; ++i) dst[i] = z;
    }
    if (tid < 32) sTid[tid] = (rs + tid < lim) ? tidl[moe * NB + rs + tid] : -1;
  }
  __syncthreads();

  int wid = tid >> 5, lane = tid & 31;
  int m = lane & 15, hf = lane >> 4;
  int mat = moe * NE + e;

  // GEMM1: X(32x1024) @ W1(1024x2048) + b1, relu -> sH (f16)
  {
    const _Float16* Wm = w1f + (size_t)mat * (128 * 32 * 512);
    const float* bm = b1 + (size_t)mat * NHID;
    const _Float16* a0p = sA + m * ND + hf * 8;
    const _Float16* a1p = sA + (16 + m) * ND + hf * 8;
    for (int g = 0; g < 8; ++g) {
      int nt0 = wid * 16 + g * 2;
      v8f C00 = {}, C01 = {}, C10 = {}, C11 = {};
      const _Float16* bp0 = Wm + (size_t)(nt0) * 32 * 512 + lane * 16;
      const _Float16* bp1 = Wm + (size_t)(nt0 + 1) * 32 * 512 + lane * 16;
#pragma unroll 4
      for (int kt = 0; kt < 32; ++kt) {
        v16h A0 = ldA(a0p + kt * 32);
        v16h A1 = ldA(a1p + kt * 32);
        v16h B0 = ldB(bp0 + (size_t)kt * 512);
        v16h B1 = ldB(bp1 + (size_t)kt * 512);
        C00 = wmma_f16(A0, B0, C00);
        C01 = wmma_f16(A0, B1, C01);
        C10 = wmma_f16(A1, B0, C10);
        C11 = wmma_f16(A1, B1, C11);
      }
#pragma unroll
      for (int q = 0; q < 4; ++q) {
        v8f Cq = (q == 0) ? C00 : (q == 1) ? C01 : (q == 2) ? C10 : C11;
        int nt = nt0 + (q & 1);
        int mb = (q >> 1) * 16;
        int n = nt * 16 + m;
        float bias = bm[n];
#pragma unroll
        for (int r = 0; r < 8; ++r) {
          float v = Cq[r] + bias;
          v = v > 0.f ? v : 0.f;
          sH[(mb + r + 8 * hf) * NHID + n] = (_Float16)v;
        }
      }
    }
  }
  __syncthreads();

  // GEMM2: H(32x2048) @ W2(2048x1024) + b2 -> scatter rows to mout (gate == 1.0)
  {
    const _Float16* Wm = w2f + (size_t)mat * (64 * 64 * 512);
    const float* bm = b2 + (size_t)mat * ND;
    const _Float16* a0p = sH + m * NHID + hf * 8;
    const _Float16* a1p = sH + (16 + m) * NHID + hf * 8;
    for (int g = 0; g < 4; ++g) {
      int nt0 = wid * 8 + g * 2;
      v8f C00 = {}, C01 = {}, C10 = {}, C11 = {};
      const _Float16* bp0 = Wm + (size_t)(nt0) * 64 * 512 + lane * 16;
      const _Float16* bp1 = Wm + (size_t)(nt0 + 1) * 64 * 512 + lane * 16;
#pragma unroll 4
      for (int kt = 0; kt < 64; ++kt) {
        v16h A0 = ldA(a0p + kt * 32);
        v16h A1 = ldA(a1p + kt * 32);
        v16h B0 = ldB(bp0 + (size_t)kt * 512);
        v16h B1 = ldB(bp1 + (size_t)kt * 512);
        C00 = wmma_f16(A0, B0, C00);
        C01 = wmma_f16(A0, B1, C01);
        C10 = wmma_f16(A1, B0, C10);
        C11 = wmma_f16(A1, B1, C11);
      }
#pragma unroll
      for (int q = 0; q < 4; ++q) {
        v8f Cq = (q == 0) ? C00 : (q == 1) ? C01 : (q == 2) ? C10 : C11;
        int nt = nt0 + (q & 1);
        int mb = (q >> 1) * 16;
        int n = nt * 16 + m;
        float bias = bm[n];
#pragma unroll
        for (int r = 0; r < 8; ++r) {
          int rowl = mb + r + 8 * hf;
          int tk = sTid[rowl];
          if (tk >= 0)
            mout[((size_t)moe * NB + tk) * ND + n] = Cq[r] + bias;
        }
      }
    }
  }
}

// ---------------- final: poly of 5 MoE fields + sigmoid ----------------
__global__ __launch_bounds__(256) void final_k(const float* __restrict__ x,
                                               const float* __restrict__ mout,
                                               float* __restrict__ out) {
  size_t idx = (size_t)blockIdx.x * 256 + threadIdx.x;
  int b = (int)(idx / (TT * ND));
  int hw = (int)(idx % ND);      // layout b*10240 + t*1024 + hw
  float xv = x[idx];
  size_t base = (size_t)b * ND + hw;
  float f0 = mout[0 * (size_t)NB * ND + base];  // transform
  float f1 = mout[1 * (size_t)NB * ND + base];  // add
  float f2 = mout[2 * (size_t)NB * ND + base];  // quad
  float f3 = mout[3 * (size_t)NB * ND + base];  // cubic
  float f4 = mout[4 * (size_t)NB * ND + base];  // fourth
  float i2 = xv * xv;
  float p = i2 * i2 * f4 + i2 * xv * f3 + i2 * f2 + xv * f0 + f1;
  out[idx] = 1.f / (1.f + __expf(-p));
}

// ---------------- launch ----------------
extern "C" void kernel_launch(void* const* d_in, const int* in_sizes, int n_in,
                              void* d_out, int out_size, void* d_ws, size_t ws_size,
                              hipStream_t stream) {
  (void)in_sizes; (void)n_in; (void)ws_size;
  const float* in  = (const float*)d_in[0];
  const float* cw1 = (const float*)d_in[1];
  const float* cb1 = (const float*)d_in[2];
  const float* cw2 = (const float*)d_in[3];
  const float* cb2 = (const float*)d_in[4];
  const float* wg  = (const float*)d_in[5];
  const float* w1  = (const float*)d_in[6];
  const float* b1  = (const float*)d_in[7];
  const float* w2  = (const float*)d_in[8];
  const float* b2  = (const float*)d_in[9];

  char* ws = (char*)d_ws;
  float*    tok    = (float*)(ws + O_TOK);
  _Float16* tokh   = (_Float16*)(ws + O_TOKH);
  _Float16* w1f    = (_Float16*)(ws + O_W1F);
  _Float16* w2f    = (_Float16*)(ws + O_W2F);
  float*    mout   = (float*)(ws + O_MOUT);
  int*      sel    = (int*)(ws + O_SEL);
  int*      tidl   = (int*)(ws + O_TIDL);
  int*      counts = (int*)(ws + O_CNT);
  int*      offs   = (int*)(ws + O_OFF);
  int*      curs   = (int*)(ws + O_CUR);
  int*      nblk   = (int*)(ws + O_NBLK);
  int2*     tiles  = (int2*)(ws + O_TILES);

  hipFuncSetAttribute((const void*)conv_fused_k,
                      hipFuncAttributeMaxDynamicSharedMemorySize, CONV_SMEM);
  hipFuncSetAttribute((const void*)moe_gemm_k,
                      hipFuncAttributeMaxDynamicSharedMemorySize, GEMM_SMEM);

  hipMemsetAsync(ws + O_CNT, 0, 264, stream);

  // weight repack to f16 WMMA fragments (20 mats each)
  prep_frags<<<20480, 256, 0, stream>>>(w1, w1f, ND, NHID, 32, 128);
  prep_frags<<<20480, 256, 0, stream>>>(w2, w2f, NHID, ND, 64, 64);

  conv_fused_k<<<2048, 256, CONV_SMEM, stream>>>(in, cw1, cb1, cw2, cb2, tok, tokh);

  gating_k<<<dim3(128, NMOE), 256, 0, stream>>>(tok, wg, sel, counts);
  setup_k<<<NMOE, 32, 0, stream>>>(counts, offs, tiles, nblk);
  loss_k<<<1, 32, 0, stream>>>(counts, (float*)d_out, out_size);
  scatter_k<<<dim3(4, NMOE), 256, 0, stream>>>(sel, offs, curs, tidl);

  moe_gemm_k<<<dim3(MAXT, NMOE), 256, GEMM_SMEM, stream>>>(
      tokh, w1f, w2f, b1, b2, tidl, counts, offs, tiles, nblk, mout);

  final_k<<<40960, 256, 0, stream>>>(in, mout, (float*)d_out);
}
